// WaltzComposer_53197464928583
// MI455X (gfx1250) — matrix operand order repaired and checked by reading, not compiled, not run on hardware
//
#include <hip/hip_runtime.h>
#include <hip/hip_bf16.h>
#include <stdint.h>

// Problem dims (fixed by reference)
#define B_  64
#define T_  256
#define V_  4096
#define E_  512
#define H_  1024
#define G3  (3*H_)
#define BT  (B_*T_)

// LDS tile row pitch in bf16 elements: 32 data + 8 pad = 80 bytes.
// Row-to-row bank offset = 20 banks -> 16B ds_load_b128 reads are conflict-free.
#define LDS_PITCH 40

typedef __attribute__((ext_vector_type(16))) __bf16 bf16x16;
typedef __attribute__((ext_vector_type(8)))  __bf16 bf16x8;
typedef __attribute__((ext_vector_type(8)))  float  f32x8;

__device__ __forceinline__ unsigned short f2bf(float f) {
  unsigned int u = __float_as_uint(f);
  u += 0x7fffu + ((u >> 16) & 1u);        // round-to-nearest-even
  return (unsigned short)(u >> 16);
}
__device__ __forceinline__ float bf2f(unsigned short s) {
  return __uint_as_float(((unsigned int)s) << 16);
}

__device__ __forceinline__ bf16x16 combine8(bf16x8 lo, bf16x8 hi) {
  union { bf16x16 v; bf16x8 h[2]; } u;
  u.h[0] = lo; u.h[1] = hi;
  return u.v;
}

// ---- gfx1250 async copy: global memory -> LDS, 16 bytes per lane ----
// VDST VGPR carries the LDS byte address (low 32 bits of the flat shared ptr),
// VADDR carries the 64-bit global address. Tracked by ASYNCcnt.
__device__ __forceinline__ void async_copy_b128(const unsigned short* lds_ptr,
                                                const unsigned short* gptr) {
  unsigned int lds_off = (unsigned int)(uintptr_t)lds_ptr;  // LDS aperture: addr[31:0]
  asm volatile("global_load_async_to_lds_b128 %0, %1, off"
               :: "v"(lds_off), "v"(gptr)
               : "memory");
}
__device__ __forceinline__ void wait_asynccnt_le4() {
  asm volatile("s_wait_asynccnt 0x4" ::: "memory");
}
__device__ __forceinline__ void wait_asynccnt_0() {
  asm volatile("s_wait_asynccnt 0x0" ::: "memory");
}

// ---- WMMA bf16 16x16x32 fragment loaders from global (per CDNA5 ISA) ----
// A (16x32 row-major): lanes 0-15: row=lane, K 0..7 / 16..23; lanes 16-31: K 8..15 / 24..31
__device__ __forceinline__ bf16x16 load_frag_a(const unsigned short* __restrict__ A,
                                               int lda, int mbase, int kbase, int lane) {
  int row = mbase + (lane & 15);
  int kh  = (lane >> 4) << 3;  // 0 or 8
  const bf16x8* p0 = (const bf16x8*)(A + (size_t)row * lda + kbase + kh);
  const bf16x8* p1 = (const bf16x8*)(A + (size_t)row * lda + kbase + 16 + kh);
  return combine8(*p0, *p1);
}
// B (32x16): C = A * W^T with W row-major [N,K]; lane holds column n = lane&15;
// lanes 0-15 hold K 0..7 / 16..23, lanes 16-31 hold K 8..15 / 24..31 (same as A).
__device__ __forceinline__ bf16x16 load_frag_b(const unsigned short* __restrict__ W,
                                               int ldb, int nbase, int kbase, int lane) {
  int col = nbase + (lane & 15);
  int ko  = (lane >> 4) << 3;  // 0 or 8
  const bf16x8* p0 = (const bf16x8*)(W + (size_t)col * ldb + kbase + ko);
  const bf16x8* p1 = (const bf16x8*)(W + (size_t)col * ldb + kbase + ko + 16);
  return combine8(*p0, *p1);
}

#define WMMA_BF16(a, b, c) \
  __builtin_amdgcn_wmma_f32_16x16x32_bf16(false, (a), false, (b), (short)0, (c), false, false)

// ---------------------------------------------------------------------------
// Big GEMM: C[M,N] = A[M,K](bf16) * W[N,K]^T(bf16) + bias[N], optional ReLU.
// Block = 8 waves (2 M x 4 N), wave tile 64x32 -> block tile 128x128.
// K pipelined in steps of 32 through double-buffered LDS filled with
// global_load_async_to_lds_b128 (ASYNCcnt), fragments read back via ds_load.
// grid.x = N/128, grid.y = M/128 (all dims divide evenly here).
// ---------------------------------------------------------------------------
__global__ __launch_bounds__(256)
void wz_gemm_bias_kernel(const unsigned short* __restrict__ A, int lda,
                         const unsigned short* __restrict__ W, int ldb,
                         const float* __restrict__ bias,
                         float* __restrict__ C, int ldc,
                         int K, int relu) {
  __shared__ __align__(16) unsigned short smA[2][128 * LDS_PITCH];  // 2 x 10 KB
  __shared__ __align__(16) unsigned short smB[2][128 * LDS_PITCH];  // 2 x 10 KB

  const int tid  = threadIdx.x;
  const int lane = tid & 31;
  const int wave = tid >> 5;
  const int wm   = (wave >> 2) * 64;   // wave M offset within block tile
  const int wn   = (wave & 3) * 32;    // wave N offset within block tile
  const int mblk = blockIdx.y * 128;
  const int nblk = blockIdx.x * 128;

  // Cooperative tile fill: 128 rows x 64B, chunk = 16B. 512 chunks per matrix,
  // 256 threads -> 2 chunks each for A and for B = 4 async issues per thread.
  auto issue_tile = [&](int buf, int k) {
#pragma unroll
    for (int i = 0; i < 2; ++i) {
      const int c   = i * 256 + tid;
      const int row = c >> 2;
      const int ce  = (c & 3) * 8;  // chunk offset in elements
      async_copy_b128(&smA[buf][row * LDS_PITCH + ce],
                      A + (size_t)(mblk + row) * lda + k + ce);
      async_copy_b128(&smB[buf][row * LDS_PITCH + ce],
                      W + (size_t)(nblk + row) * ldb + k + ce);
    }
  };

  f32x8 acc[4][2] = {};

  issue_tile(0, 0);
  int buf = 0;
  for (int k = 0; k < K; k += 32) {
    const bool more = (k + 32 < K);
    if (more) {
      issue_tile(buf ^ 1, k + 32);   // prefetch next tile into other buffer
      wait_asynccnt_le4();           // wait only for *previous* buffer's 4 copies
    } else {
      wait_asynccnt_0();
    }
    __syncthreads();                 // all waves' copies of `buf` now visible

    const unsigned short* As = smA[buf];
    const unsigned short* Bs = smB[buf];
    const int kh = (lane >> 4) << 3;   // 0 or 8
    const int ko = (lane >> 4) << 3;   // 0 or 8
    const int rl = lane & 15;

    bf16x16 a[4], b[2];
#pragma unroll
    for (int mi = 0; mi < 4; ++mi) {
      const int lrow = (wm + mi * 16 + rl) * LDS_PITCH;
      a[mi] = combine8(*(const bf16x8*)(As + lrow + kh),
                       *(const bf16x8*)(As + lrow + 16 + kh));
    }
#pragma unroll
    for (int ni = 0; ni < 2; ++ni) {
      const int lcol = (wn + ni * 16 + rl) * LDS_PITCH;
      b[ni] = combine8(*(const bf16x8*)(Bs + lcol + ko),
                       *(const bf16x8*)(Bs + lcol + ko + 16));
    }

    acc[0][0] = WMMA_BF16(a[0], b[0], acc[0][0]);
    acc[1][0] = WMMA_BF16(a[1], b[0], acc[1][0]);
    acc[2][0] = WMMA_BF16(a[2], b[0], acc[2][0]);
    acc[3][0] = WMMA_BF16(a[3], b[0], acc[3][0]);
    acc[0][1] = WMMA_BF16(a[0], b[1], acc[0][1]);
    acc[1][1] = WMMA_BF16(a[1], b[1], acc[1][1]);
    acc[2][1] = WMMA_BF16(a[2], b[1], acc[2][1]);
    acc[3][1] = WMMA_BF16(a[3], b[1], acc[3][1]);

    __syncthreads();                 // everyone done reading `buf` before it is
    buf ^= 1;                        // overwritten by next iteration's issues
  }

  const int hi8  = (lane >> 4) * 8;
  const int colb = lane & 15;
#pragma unroll
  for (int mi = 0; mi < 4; ++mi) {
#pragma unroll
    for (int ni = 0; ni < 2; ++ni) {
      const int col = nblk + wn + ni * 16 + colb;
      const float bv = bias[col];
#pragma unroll
      for (int i = 0; i < 8; ++i) {
        const int row = mblk + wm + mi * 16 + hi8 + i;
        float v = acc[mi][ni][i] + bv;
        if (relu) v = fmaxf(v, 0.0f);
        C[(size_t)row * ldc + col] = v;
      }
    }
  }
}

// ---------------------------------------------------------------------------
// Fused GRU recurrence step:  hW = h_prev @ W_hh^T (+b_hh) via WMMA, then gate
// nonlinearity + state update, one launch per timestep. Block = 4 waves; wave
// owns batch rows [16w,16w+16); block owns 16 hidden cols; computing the same
// 16x16 tile for the r/z/n gate slices puts r,z,n for a given (b,j) in the
// same lane + frag index -> pure per-lane register nonlinearity. grid = H/16.
// Direct global loads: latency-bound step, W_hh (6 MB bf16) is L2-resident.
// ---------------------------------------------------------------------------
__global__ __launch_bounds__(128)
void wz_gru_step_kernel(const float* __restrict__ xw,           // [BT,3H], b_ih included
                        const unsigned short* __restrict__ Whh, // [3H,H] bf16
                        const float* __restrict__ b_hh,         // [3H]
                        const unsigned short* __restrict__ h_in, // [B,H] bf16
                        unsigned short* __restrict__ h_out,      // [B,H] bf16
                        unsigned short* __restrict__ h_seq,      // [B,T,H] bf16
                        int t) {
  const int lane  = threadIdx.x & 31;
  const int mbase = (threadIdx.x >> 5) * 16;  // batch-row tile
  const int ncol  = blockIdx.x * 16;          // hidden-col tile

  f32x8 cr = {}, cz = {}, cn = {};
  for (int k = 0; k < H_; k += 32) {
    bf16x16 a  = load_frag_a(h_in, H_, mbase, k, lane);
    bf16x16 br = load_frag_b(Whh, H_, ncol,          k, lane);
    bf16x16 bz = load_frag_b(Whh, H_, ncol + H_,     k, lane);
    bf16x16 bn = load_frag_b(Whh, H_, ncol + 2 * H_, k, lane);
    cr = WMMA_BF16(a, br, cr);
    cz = WMMA_BF16(a, bz, cz);
    cn = WMMA_BF16(a, bn, cn);
  }

  const int hi8 = (lane >> 4) * 8;
  const int j   = ncol + (lane & 15);
  const float bhr = b_hh[j];
  const float bhz = b_hh[H_ + j];
  const float bhn = b_hh[2 * H_ + j];
#pragma unroll
  for (int i = 0; i < 8; ++i) {
    const int b = mbase + hi8 + i;
    const float* xrow = xw + ((size_t)b * T_ + t) * G3;
    const float r = 1.0f / (1.0f + __expf(-(xrow[j]        + cr[i] + bhr)));
    const float z = 1.0f / (1.0f + __expf(-(xrow[H_ + j]   + cz[i] + bhz)));
    const float n = tanhf(xrow[2 * H_ + j] + r * (cn[i] + bhn));
    const float hold = bf2f(h_in[(size_t)b * H_ + j]);
    const float hnew = (1.0f - z) * n + z * hold;
    const unsigned short hb = f2bf(hnew);
    h_out[(size_t)b * H_ + j] = hb;
    h_seq[((size_t)b * T_ + t) * H_ + j] = hb;
  }
}

// ---------------------------------------------------------------------------
// Helpers: embedding gather (f32 -> bf16), weight f32->bf16, bf16 zero-fill
// ---------------------------------------------------------------------------
__global__ __launch_bounds__(128)
void wz_embed_kernel(const int* __restrict__ tok, const float* __restrict__ emb,
                     unsigned short* __restrict__ x) {
  const int n  = blockIdx.x;
  const int id = tok[n];
  const float* src = emb + (size_t)id * E_;
  unsigned short* dst = x + (size_t)n * E_;
  for (int e = threadIdx.x; e < E_; e += blockDim.x) dst[e] = f2bf(src[e]);
}

__global__ void wz_f32_to_bf16_kernel(const float* __restrict__ src,
                                      unsigned short* __restrict__ dst, int n) {
  const int i = blockIdx.x * blockDim.x + threadIdx.x;
  if (i < n) dst[i] = f2bf(src[i]);
}

__global__ void wz_zero_bf16_kernel(unsigned short* __restrict__ p, int n) {
  const int i = blockIdx.x * blockDim.x + threadIdx.x;
  if (i < n) p[i] = 0;
}

// ---------------------------------------------------------------------------
extern "C" void kernel_launch(void* const* d_in, const int* in_sizes, int n_in,
                              void* d_out, int out_size, void* d_ws, size_t ws_size,
                              hipStream_t stream) {
  (void)in_sizes; (void)n_in; (void)out_size; (void)ws_size;

  const int*   tok  = (const int*)d_in[0];
  const float* emb  = (const float*)d_in[1];
  const float* Wih0 = (const float*)d_in[2];
  const float* Whh0 = (const float*)d_in[3];
  const float* bih0 = (const float*)d_in[4];
  const float* bhh0 = (const float*)d_in[5];
  const float* Wih1 = (const float*)d_in[6];
  const float* Whh1 = (const float*)d_in[7];
  const float* bih1 = (const float*)d_in[8];
  const float* bhh1 = (const float*)d_in[9];
  const float* Wfc  = (const float*)d_in[10];
  const float* bfc  = (const float*)d_in[11];
  float* out = (float*)d_out;

  // ---- scratch layout (~301 MB) ----
  char* ws = (char*)d_ws;
  size_t off = 0;
  auto alloc = [&](size_t bytes) -> char* {
    char* p = ws + off;
    off += (bytes + 255) & ~(size_t)255;
    return p;
  };
  unsigned short* x_bf   = (unsigned short*)alloc((size_t)BT * E_ * 2);
  unsigned short* wih0_b = (unsigned short*)alloc((size_t)G3 * E_ * 2);
  unsigned short* whh0_b = (unsigned short*)alloc((size_t)G3 * H_ * 2);
  unsigned short* wih1_b = (unsigned short*)alloc((size_t)G3 * H_ * 2);
  unsigned short* whh1_b = (unsigned short*)alloc((size_t)G3 * H_ * 2);
  unsigned short* wfc_b  = (unsigned short*)alloc((size_t)V_ * H_ * 2);
  float*          xwbuf  = (float*)alloc((size_t)BT * G3 * 4);  // reused L0 then L1
  unsigned short* h1seq  = (unsigned short*)alloc((size_t)BT * H_ * 2);
  unsigned short* h2seq  = (unsigned short*)alloc((size_t)BT * H_ * 2);
  unsigned short* hping  = (unsigned short*)alloc((size_t)B_ * H_ * 2);
  unsigned short* hpong  = (unsigned short*)alloc((size_t)B_ * H_ * 2);

  // ---- one-time conversions ----
  auto conv = [&](const float* s, unsigned short* d, int n) {
    wz_f32_to_bf16_kernel<<<(n + 255) / 256, 256, 0, stream>>>(s, d, n);
  };
  conv(Wih0, wih0_b, G3 * E_);
  conv(Whh0, whh0_b, G3 * H_);
  conv(Wih1, wih1_b, G3 * H_);
  conv(Whh1, whh1_b, G3 * H_);
  conv(Wfc,  wfc_b,  V_ * H_);

  wz_embed_kernel<<<BT, 128, 0, stream>>>(tok, emb, x_bf);

  // ---- layer 0: input projection (one big GEMM) + 256 fused recurrent steps
  wz_gemm_bias_kernel<<<dim3(G3 / 128, BT / 128), 256, 0, stream>>>(
      x_bf, E_, wih0_b, E_, bih0, xwbuf, G3, E_, 0);
  wz_zero_bf16_kernel<<<(B_ * H_ + 255) / 256, 256, 0, stream>>>(hping, B_ * H_);
  {
    unsigned short *hin = hping, *hout = hpong;
    for (int t = 0; t < T_; ++t) {
      wz_gru_step_kernel<<<H_ / 16, 128, 0, stream>>>(xwbuf, whh0_b, bhh0,
                                                      hin, hout, h1seq, t);
      unsigned short* tmp = hin; hin = hout; hout = tmp;
    }
  }

  // ---- layer 1 ----
  wz_gemm_bias_kernel<<<dim3(G3 / 128, BT / 128), 256, 0, stream>>>(
      h1seq, H_, wih1_b, H_, bih1, xwbuf, G3, H_, 0);
  wz_zero_bf16_kernel<<<(B_ * H_ + 255) / 256, 256, 0, stream>>>(hping, B_ * H_);
  {
    unsigned short *hin = hping, *hout = hpong;
    for (int t = 0; t < T_; ++t) {
      wz_gru_step_kernel<<<H_ / 16, 128, 0, stream>>>(xwbuf, whh1_b, bhh1,
                                                      hin, hout, h2seq, t);
      unsigned short* tmp = hin; hin = hout; hout = tmp;
    }
  }

  // ---- FC + ReLU -> logits [B,T,V] f32 ----
  wz_gemm_bias_kernel<<<dim3(V_ / 128, BT / 128), 256, 0, stream>>>(
      h2seq, H_, wfc_b, H_, bfc, out, V_, H_, 1);
}
